// EncoderMeshBlock_54640573939785
// MI455X (gfx1250) — compile-verified, matrix operand order checked by reading, not run on hardware
//
#include <hip/hip_runtime.h>

// ---------------------------------------------------------------------------
// EncoderMeshBlock for MI455X (gfx1250, wave32).
// Dense matmuls -> v_wmma_f32_16x16x32_f16 (fp16 in, fp32 accum).
// Gathers staged in LDS (ds_load_b128 fragments per ISA A/B layouts).
// Scatter-mean -> hardware global_atomic_add_f32 (unsafeAtomicAdd).
// BN stats -> block reduce + FP atomics.
// NOTE: float atomics make the result deterministic only up to FP reduction
// order (standard for segment-sum scatter kernels).
// ---------------------------------------------------------------------------

typedef __attribute__((ext_vector_type(16))) _Float16 v16h;
typedef __attribute__((ext_vector_type(8)))  _Float16 v8h;
typedef __attribute__((ext_vector_type(8)))  float    v8f;

static const int V_N   = 150000;
static const int F_N   = 300000;
static const int KF    = 16;
static const int MIDC  = 128;   // MID = 4*GROWTH

// Hardware FP32 atomic add (global_atomic_add_f32), avoiding any CAS fallback.
__device__ __forceinline__ void atomic_add_f32(float* p, float v)
{
    unsafeAtomicAdd(p, v);
}

// ---------------------------------------------------------------------------
// Unified WMMA GEMM.
//   Y[m, 0..Co) = sum_{i<nmat} A[idx(m,i), 0..Ci) @ W_i   (fp32 out)
//   idx(m,i) = gidx ? gidx[m*nmat+i] : m
// A is fp16 with row stride lda; Wt is fp16, layout [nmat][Co][Ci] (N-major,
// so each lane's B fragment is two contiguous b128 chunks).
// Block = 256 threads = 8 waves; one 16-row M tile per block; wave w owns
// output columns [16w, 16w+16).
// ---------------------------------------------------------------------------
__global__ __launch_bounds__(256)
void gemm_wmma(const _Float16* __restrict__ A, int lda, int Ci,
               const int* __restrict__ gidx, int nmat,
               const _Float16* __restrict__ Wt, int Co,
               float* __restrict__ Y, int Mrows)
{
    __shared__ __align__(16) _Float16 As[16][136];   // 272B row stride: bank-staggered

    const int m0   = blockIdx.x * 16;
    const int lane = threadIdx.x & 31;
    const int wave = threadIdx.x >> 5;
    const int gr   = threadIdx.x >> 4;   // gather row   0..15
    const int gc   = threadIdx.x & 15;   // gather chunk 0..15 (16B each)
    const bool wactive = (wave * 16) < Co;
    const int n  = wave * 16 + (lane & 15);          // global output column
    const int kb = (lane & 16) ? 8 : 0;              // A/B fragment K base
    const int mm = lane & 15;                        // A fragment row

    v8f acc = {};
    for (int i = 0; i < nmat; ++i) {
        __syncthreads();                 // protect As from previous iteration reads
        if (gc * 8 < Ci) {
            int m = m0 + gr;
            if (m < Mrows) {
                int src = gidx ? gidx[m * nmat + i] : m;
                *(v8h*)(&As[gr][gc * 8]) =
                    *(const v8h*)(A + (size_t)src * lda + gc * 8);
            }
        }
        __syncthreads();
        if (wactive) {
            const _Float16* Wm = Wt + ((size_t)i * Co + n) * Ci;
            __builtin_prefetch(Wm, 0, 1);            // global_prefetch_b8
            for (int kt = 0; kt < Ci; kt += 32) {
                v8h alo = *(const v8h*)(&As[mm][kt + kb]);
                v8h ahi = *(const v8h*)(&As[mm][kt + kb + 16]);
                v16h a  = __builtin_shufflevector(alo, ahi,
                           0,1,2,3,4,5,6,7,8,9,10,11,12,13,14,15);
                v8h blo = *(const v8h*)(Wm + kt + kb);
                v8h bhi = *(const v8h*)(Wm + kt + kb + 16);
                v16h b  = __builtin_shufflevector(blo, bhi,
                           0,1,2,3,4,5,6,7,8,9,10,11,12,13,14,15);
                acc = __builtin_amdgcn_wmma_f32_16x16x32_f16(
                          false, a, false, b, (short)0, acc, false, false);
            }
        }
    }
    if (wactive) {
        const int mb = (lane & 16) ? 8 : 0;          // D layout: M = r + 8*(lane>=16)
#pragma unroll
        for (int r = 0; r < 8; ++r) {
            int m = m0 + mb + r;
            if (m < Mrows) Y[(size_t)m * Co + n] = acc[r];
        }
    }
}

// --- BN column statistics: partial sums/sumsq per channel -> atomics ---------
__global__ __launch_bounds__(256)
void colstats(const float* __restrict__ y, int nrows, int C,
              float* __restrict__ sums /* [2C], pre-zeroed */)
{
    __shared__ float rs[256], rq[256];
    const int c   = threadIdx.x % C;
    const int rr  = threadIdx.x / C;
    const int tpc = 256 / C;
    float s = 0.f, q = 0.f;
    for (int r = blockIdx.x * tpc + rr; r < nrows; r += gridDim.x * tpc) {
        float v = y[(size_t)r * C + c];
        s += v; q += v * v;
    }
    rs[threadIdx.x] = s; rq[threadIdx.x] = q;
    __syncthreads();
    if (rr == 0) {
        for (int j = 1; j < tpc; ++j) { s += rs[c + j * C]; q += rq[c + j * C]; }
        atomic_add_f32(&sums[c], s);
        atomic_add_f32(&sums[C + c], q);
    }
}

__global__ void bnfinalize(const float* __restrict__ sums,
                           const float* __restrict__ gamma,
                           const float* __restrict__ beta,
                           float invN, int C,
                           float* __restrict__ scale, float* __restrict__ shift)
{
    int c = threadIdx.x;
    if (c < C) {
        float m   = sums[c] * invN;
        float var = sums[C + c] * invN - m * m;
        float sc  = gamma[c] * rsqrtf(var + 1e-5f);
        scale[c] = sc;
        shift[c] = beta[c] - m * sc;
    }
}

// --- f2v front half: z = relu(bn(y)) * (coeff @ Wd); scatter-add to vertices -
__global__ __launch_bounds__(256)
void fz_scatter(const float* __restrict__ Y, const float* __restrict__ coeff,
                const float* __restrict__ Wd, const float* __restrict__ scale,
                const float* __restrict__ shift, const int* __restrict__ vt,
                float* __restrict__ acc, int F_)
{
    __shared__ float sWd[KF * MIDC];
    for (int i = threadIdx.x; i < KF * MIDC; i += 256) sWd[i] = Wd[i];
    __syncthreads();
    int idx = blockIdx.x * 256 + threadIdx.x;
    if (idx >= F_ * MIDC) return;
    int f = idx >> 7;
    int c = idx & 127;
    float s = 0.f;
#pragma unroll
    for (int k = 0; k < KF; ++k) s = fmaf(coeff[f * KF + k], sWd[k * MIDC + c], s);
    float h = fmaxf(fmaf(Y[idx], scale[c], shift[c]), 0.f);
    float z = h * s;
    int b = f * 3;
    atomic_add_f32(&acc[(size_t)vt[b + 0] * MIDC + c], z);
    atomic_add_f32(&acc[(size_t)vt[b + 1] * MIDC + c], z);
    atomic_add_f32(&acc[(size_t)vt[b + 2] * MIDC + c], z);
}

// --- elementwise helpers -----------------------------------------------------
__global__ void nfinv_kernel(const int* __restrict__ nf, float* __restrict__ nfi, int n)
{
    int i = blockIdx.x * blockDim.x + threadIdx.x;
    if (i < n) { int c = nf[i]; if (c < 1) c = 1; nfi[i] = 1.0f / (float)c; }
}

__global__ void init_xh(const float* __restrict__ inp, _Float16* __restrict__ XH, int V_)
{
    int i = blockIdx.x * blockDim.x + threadIdx.x;
    if (i >= V_ * 64) return;
    int v = i >> 6, c = i & 63;
    XH[(size_t)v * 128 + c] = (_Float16)inp[i];
}

__global__ void convWt(const float* __restrict__ W, _Float16* __restrict__ Wt,
                       int Ci, int Co)
{
    int i = blockIdx.x * blockDim.x + threadIdx.x;
    if (i >= Ci * Co) return;
    int k = i / Co, co = i % Co;
    Wt[co * Ci + k] = (_Float16)W[i];
}

__global__ void prep_a16(const float* __restrict__ acc, const float* __restrict__ nfi,
                         _Float16* __restrict__ a16, int n)
{
    int i = blockIdx.x * blockDim.x + threadIdx.x;
    if (i < n) a16[i] = (_Float16)(acc[i] * nfi[i >> 7]);
}

__global__ void bnrelu_f16(const float* __restrict__ y, const float* __restrict__ scale,
                           const float* __restrict__ shift, _Float16* __restrict__ out, int n)
{
    int i = blockIdx.x * blockDim.x + threadIdx.x;
    if (i >= n) return;
    int c = i & 127;
    out[i] = (_Float16)fmaxf(fmaf(y[i], scale[c], shift[c]), 0.f);
}

__global__ void bnrelu_concat(const float* __restrict__ y, const float* __restrict__ scale,
                              const float* __restrict__ shift, _Float16* __restrict__ XH,
                              int V_, int col0)
{
    int i = blockIdx.x * blockDim.x + threadIdx.x;
    if (i >= V_ * 32) return;
    int v = i >> 5, c = i & 31;
    float h = fmaxf(fmaf(y[i], scale[c], shift[c]), 0.f);
    XH[(size_t)v * 128 + col0 + c] = (_Float16)h;
}

__global__ void bnrelu_out(const float* __restrict__ y, const float* __restrict__ scale,
                           const float* __restrict__ shift, float* __restrict__ out, int n)
{
    int i = blockIdx.x * blockDim.x + threadIdx.x;
    if (i >= n) return;
    int c = i & 127;
    out[i] = fmaxf(fmaf(y[i], scale[c], shift[c]), 0.f);
}

// ---------------------------------------------------------------------------
extern "C" void kernel_launch(void* const* d_in, const int* in_sizes, int n_in,
                              void* d_out, int out_size, void* d_ws, size_t ws_size,
                              hipStream_t stream)
{
    (void)in_sizes; (void)n_in; (void)out_size; (void)ws_size;
    const int V_ = V_N, F_ = F_N;
    const float* inputs = (const float*)d_in[0];
    const int*   face   = (const int*)d_in[2];
    const int*   nfcnt  = (const int*)d_in[3];
    const int*   vt     = (const int*)d_in[4];
    const float* coeff  = (const float*)d_in[5];
    auto P = [&](int i) -> const float* { return (const float*)d_in[7 + i]; };

    // ---- workspace carve (≈310 MB; all buffers written before read) ----
    char* wp = (char*)d_ws;
    auto carve = [&](size_t bytes) -> void* {
        void* p = (void*)wp; wp += (bytes + 255) & ~(size_t)255; return p;
    };
    _Float16* XH   = (_Float16*)carve((size_t)V_ * 128 * 2); // feature map (concat in place)
    _Float16* A16  = (_Float16*)carve((size_t)V_ * 128 * 2); // fp16 GEMM A staging
    float*    ACC  = (float*)   carve((size_t)V_ * 128 * 4); // scatter accumulator / vertex fp32
    float*    YF   = (float*)   carve((size_t)F_ * 128 * 4); // face buffer (reused)
    float*    NFI  = (float*)   carve((size_t)V_ * 4);
    float*    SUMS = (float*)   carve(256 * 4);
    float*    SCALE= (float*)   carve(128 * 4);
    float*    SHIFT= (float*)   carve(128 * 4);
    _Float16* WT1[2], *WP1[2], *WT2[2], *WP2[2];
    for (int n = 0; n < 2; ++n) {
        int cin = 64 + 32 * n;
        WT1[n] = (_Float16*)carve((size_t)3 * cin * 128 * 2);
        WP1[n] = (_Float16*)carve((size_t)128 * 128 * 2);
        WT2[n] = (_Float16*)carve((size_t)3 * 128 * 128 * 2);
        WP2[n] = (_Float16*)carve((size_t)32 * 128 * 2);
    }
    _Float16* TWT = (_Float16*)carve((size_t)128 * 128 * 2);

    auto conv = [&](const float* Ws, _Float16* Wd, int Ci, int Co) {
        int nEl = Ci * Co;
        convWt<<<(nEl + 255) / 256, 256, 0, stream>>>(Ws, Wd, Ci, Co);
    };
    auto bnstats = [&](const float* y, int nrows, int C,
                       const float* gamma, const float* beta) {
        hipMemsetAsync(SUMS, 0, 2 * C * sizeof(float), stream);
        colstats<<<960, 256, 0, stream>>>(y, nrows, C, SUMS);
        bnfinalize<<<1, 128, 0, stream>>>(SUMS, gamma, beta, 1.0f / nrows, C, SCALE, SHIFT);
    };

    // ---- prologue ----
    nfinv_kernel<<<(V_ + 255) / 256, 256, 0, stream>>>(nfcnt, NFI, V_);
    init_xh<<<(V_ * 64 + 255) / 256, 256, 0, stream>>>(inputs, XH, V_);

    // ---- dense blocks ----
    for (int n = 0; n < 2; ++n) {
        int base = n * 14;
        int cin  = 64 + 32 * n;
        const float *W1 = P(base+0),  *g1  = P(base+1),  *b1  = P(base+2);
        const float *Wd1= P(base+3),  *Wp1 = P(base+4),  *gf1 = P(base+5), *bf1 = P(base+6);
        const float *W2 = P(base+7),  *g2  = P(base+8),  *b2  = P(base+9);
        const float *Wd2= P(base+10), *Wp2 = P(base+11), *gf2 = P(base+12),*bf2 = P(base+13);

        for (int i = 0; i < 3; ++i)
            conv(W1 + (size_t)i * cin * 128, WT1[n] + (size_t)i * 128 * cin, cin, 128);
        conv(Wp1, WP1[n], 128, 128);
        for (int i = 0; i < 3; ++i)
            conv(W2 + (size_t)i * 128 * 128, WT2[n] + (size_t)i * 128 * 128, 128, 128);
        conv(Wp2, WP2[n], 128, 32);

        // V2VConv3d #1 : v2f (gather GEMM) -> BN -> fuzzy scale + scatter -> pointwise
        gemm_wmma<<<(F_ + 15) / 16, 256, 0, stream>>>(XH, 128, cin, face, 3, WT1[n], 128, YF, F_);
        bnstats(YF, F_, 128, g1, b1);
        hipMemsetAsync(ACC, 0, (size_t)V_ * 128 * 4, stream);
        fz_scatter<<<(F_ * 128) / 256, 256, 0, stream>>>(YF, coeff, Wd1, SCALE, SHIFT, vt, ACC, F_);
        prep_a16<<<(V_ * 128 + 255) / 256, 256, 0, stream>>>(ACC, NFI, A16, V_ * 128);
        gemm_wmma<<<(V_ + 15) / 16, 256, 0, stream>>>(A16, 128, 128, nullptr, 1, WP1[n], 128, ACC, V_);
        bnstats(ACC, V_, 128, gf1, bf1);
        bnrelu_f16<<<(V_ * 128 + 255) / 256, 256, 0, stream>>>(ACC, SCALE, SHIFT, A16, V_ * 128);

        // V2VConv3d #2 : MID -> MID -> GROWTH
        gemm_wmma<<<(F_ + 15) / 16, 256, 0, stream>>>(A16, 128, 128, face, 3, WT2[n], 128, YF, F_);
        bnstats(YF, F_, 128, g2, b2);
        hipMemsetAsync(ACC, 0, (size_t)V_ * 128 * 4, stream);
        fz_scatter<<<(F_ * 128) / 256, 256, 0, stream>>>(YF, coeff, Wd2, SCALE, SHIFT, vt, ACC, F_);
        prep_a16<<<(V_ * 128 + 255) / 256, 256, 0, stream>>>(ACC, NFI, A16, V_ * 128);
        gemm_wmma<<<(V_ + 15) / 16, 256, 0, stream>>>(A16, 128, 128, nullptr, 1, WP2[n], 32, YF, V_);
        bnstats(YF, V_, 32, gf2, bf2);
        bnrelu_concat<<<(V_ * 32 + 255) / 256, 256, 0, stream>>>(YF, SCALE, SHIFT, XH, V_, cin);
    }

    // ---- transit: per-vertex 1x1 conv + BN + ReLU -> d_out [V, 128] fp32 ----
    conv(P(28), TWT, 128, 128);
    gemm_wmma<<<(V_ + 15) / 16, 256, 0, stream>>>(XH, 128, 128, nullptr, 1, TWT, 128, ACC, V_);
    bnstats(ACC, V_, 128, P(29), P(30));
    bnrelu_out<<<(V_ * 128 + 255) / 256, 256, 0, stream>>>(ACC, SCALE, SHIFT, (float*)d_out, V_ * 128);
}
